// GCNZinc_72473278153004
// MI455X (gfx1250) — matrix operand order repaired and checked by reading, not compiled, not run on hardware
//
#include <hip/hip_runtime.h>
#include <hip/hip_bf16.h>

// CDNA5 (gfx1250) GCN inference:
//   h0 = emb[x]; h1 = relu(GCNconv(h0,W1,b1)); emb2 = GCNconv(h1,W2,b2);
//   y = emb2 @ linW + linb; out = segment_sum(y, batch)
// GEMMs on v_wmma_f32_16x16x32_bf16; A kept in bf16, W pre-packed into the
// ISA fragment layout so every operand arrives via b128 loads (no per-wave cvt).

typedef __attribute__((ext_vector_type(16))) __bf16 v16bf;
typedef __attribute__((ext_vector_type(8)))  __bf16 v8bf;
typedef __attribute__((ext_vector_type(8)))  float  v8f;

union V16 { v16bf v; v8bf h[2]; };

#define FDIM 64

// ---------------- weight pre-pack (fragment-ordered bf16) ----------------
// Wp[((nt*2+kc)*32 + lane)*16 + i] = bf16( W[(kc*32 + (lane>>4)*8 + (i&7) + (i>>3)*16)*64 + nt*16 + (lane&15)] )

__global__ void pack_w_kernel(const float* __restrict__ W1, const float* __restrict__ W2,
                              __bf16* __restrict__ Wp1, __bf16* __restrict__ Wp2) {
    int t = blockIdx.x * blockDim.x + threadIdx.x;
    if (t >= 8192) return;
    const float* W  = (t < 4096) ? W1 : W2;
    __bf16*      Wp = (t < 4096) ? Wp1 : Wp2;
    int u    = t & 4095;
    int i    = u & 15;
    int lane = (u >> 4) & 31;
    int f    = u >> 9;            // fragment id: nt*2 + kc
    int nt   = f >> 1, kc = f & 1;
    int half = lane >> 4, lmod = lane & 15;
    int k = kc * 32 + half * 8 + (i & 7) + (i >> 3) * 16;
    int n = nt * 16 + lmod;
    Wp[u] = (__bf16)W[(size_t)k * FDIM + n];
}

// ---------------- degree / embedding ----------------

__global__ void init_deg_kernel(float* __restrict__ deg, int n) {
    int i = blockIdx.x * blockDim.x + threadIdx.x;
    if (i < n) deg[i] = 1.0f;  // self-loop contributes 1 to every node's degree
}

__global__ void deg_accum_kernel(const int* __restrict__ dst, float* __restrict__ deg, int e) {
    int i = blockIdx.x * blockDim.x + threadIdx.x;
    if (i < e) atomicAdd(&deg[dst[i]], 1.0f);
}

// Abf[i,k] = bf16(emb_table[x[i],k]); lane k==0 also dinv[i] = rsqrt(deg[i])
__global__ void embed_kernel(const int* __restrict__ x,
                             const float* __restrict__ emb_table,
                             const float* __restrict__ deg,
                             __bf16* __restrict__ Abf,
                             float* __restrict__ dinv,
                             int n) {
    long long t = (long long)blockIdx.x * blockDim.x + threadIdx.x;
    long long total = (long long)n * FDIM;
    if (t >= total) return;
    int i = (int)(t >> 6);
    int k = (int)(t & (FDIM - 1));
    Abf[t] = (__bf16)emb_table[(size_t)x[i] * FDIM + k];
    if (k == 0) dinv[i] = rsqrtf(deg[i]);
}

// ---------------- WMMA GEMM: hs = (A @ W) * dinv[row]; acc = hs ----------------
// One wave -> 16-row x 64-col strip: 4 C tiles x 2 chained WMMAs (K=64).

__global__ __launch_bounds__(256)
void gemm64_scale_kernel(const __bf16* __restrict__ Abf,  // [N,64] bf16 row-major
                         const __bf16* __restrict__ Wp,   // [8][32][16] fragment-packed bf16
                         const float*  __restrict__ dinv, // [N]
                         float* __restrict__ hs,          // [N,64] out (scaled product)
                         float* __restrict__ acc,         // [N,64] out (copy = self-loop init)
                         int nTiles) {
    int wave = (int)((blockIdx.x * (long long)blockDim.x + threadIdx.x) >> 5); // wave-uniform
    int lane = threadIdx.x & 31;
    if (wave >= nTiles) return;   // whole-wave exit: EXEC stays all-1s for WMMA

    const int R    = wave << 4;
    const int half = lane >> 4;
    const int lmod = lane & 15;

    // A fragments (16x32 bf16 each): two contiguous 8-element runs per chunk.
    // lanes 0-15 (row=lmod): elems 0-7 -> K=0..7,  8-15 -> K=16..23
    // lanes 16-31:           elems 0-7 -> K=8..15, 8-15 -> K=24..31
    const __bf16* arow = Abf + (size_t)(R + lmod) * FDIM;
    V16 a0, a1;
    a0.h[0] = *(const v8bf*)(arow +      half * 8);
    a0.h[1] = *(const v8bf*)(arow + 16 + half * 8);
    a1.h[0] = *(const v8bf*)(arow + 32 + half * 8);
    a1.h[1] = *(const v8bf*)(arow + 48 + half * 8);

    // dinv for this lane's 8 output rows (C tile: lane,VGPR r -> M = r + 8*half)
    float dv[8];
#pragma unroll
    for (int r = 0; r < 8; ++r) dv[r] = dinv[R + half * 8 + r];

#pragma unroll
    for (int nt = 0; nt < 4; ++nt) {
        const v16bf b0 = *(const v16bf*)(Wp + ((size_t)(nt * 2 + 0) * 32 + lane) * 16);
        const v16bf b1 = *(const v16bf*)(Wp + ((size_t)(nt * 2 + 1) * 32 + lane) * 16);
        v8f c = {};
        c = __builtin_amdgcn_wmma_f32_16x16x32_bf16(
                false, a0.v, false, b0, (short)0, c, false, false);
        c = __builtin_amdgcn_wmma_f32_16x16x32_bf16(
                false, a1.v, false, b1, (short)0, c, false, false);
#pragma unroll
        for (int r = 0; r < 8; ++r) {
            const int row = R + half * 8 + r;
            const float v = c[r] * dv[r];
            const size_t o = (size_t)row * FDIM + nt * 16 + lmod;
            hs[o]  = v;
            acc[o] = v;
        }
    }
}

// ---------------- edge scatter: acc[dst] += hs[src] ----------------
// 64 consecutive threads share one edge -> contiguous row read + contiguous atomics.

__global__ void scatter_kernel(const int* __restrict__ src, const int* __restrict__ dst,
                               const float* __restrict__ hs, float* __restrict__ acc,
                               long long total) {
    long long t = (long long)blockIdx.x * blockDim.x + threadIdx.x;
    if (t >= total) return;
    int e = (int)(t >> 6);
    int k = (int)(t & (FDIM - 1));
    atomicAdd(&acc[(size_t)dst[e] * FDIM + k], hs[(size_t)src[e] * FDIM + k]);
}

// ---------------- finalize ----------------
// layer1: Abf = bf16(relu(acc*dinv + b))   (feeds GEMM2)
__global__ void finalize_bf16_kernel(const float* __restrict__ acc,
                                     const float* __restrict__ dinv,
                                     const float* __restrict__ b,
                                     __bf16* __restrict__ outbf, int n) {
    long long t = (long long)blockIdx.x * blockDim.x + threadIdx.x;
    long long total = (long long)n * FDIM;
    if (t >= total) return;
    int i = (int)(t >> 6);
    int k = (int)(t & (FDIM - 1));
    outbf[t] = (__bf16)fmaxf(acc[t] * dinv[i] + b[k], 0.0f);
}

// layer2: outf = acc*dinv + b   (feeds readout, f32)
__global__ void finalize_f32_kernel(const float* __restrict__ acc,
                                    const float* __restrict__ dinv,
                                    const float* __restrict__ b,
                                    float* __restrict__ outf, int n) {
    long long t = (long long)blockIdx.x * blockDim.x + threadIdx.x;
    long long total = (long long)n * FDIM;
    if (t >= total) return;
    int i = (int)(t >> 6);
    int k = (int)(t & (FDIM - 1));
    outf[t] = acc[t] * dinv[i] + b[k];
}

// ---------------- readout: out[batch[i]] += emb[i] . linW + linb ----------------

__global__ void zero_out_kernel(float* __restrict__ out, int n) {
    int i = blockIdx.x * blockDim.x + threadIdx.x;
    if (i < n) out[i] = 0.0f;
}

__global__ __launch_bounds__(256)
void readout_kernel(const float* __restrict__ emb, const float* __restrict__ linW,
                    const float* __restrict__ linb, const int* __restrict__ batch,
                    float* __restrict__ out, int n) {
    int wave = (int)((blockIdx.x * (long long)blockDim.x + threadIdx.x) >> 5);
    int lane = threadIdx.x & 31;
    if (wave >= n) return;
    const float* row = emb + (size_t)wave * FDIM;
    float s = row[lane] * linW[lane] + row[lane + 32] * linW[lane + 32];
#pragma unroll
    for (int off = 16; off > 0; off >>= 1)
        s += __shfl_xor(s, off, 32);
    if (lane == 0) atomicAdd(&out[batch[wave]], s + linb[0]);
}

// ---------------- driver ----------------

extern "C" void kernel_launch(void* const* d_in, const int* in_sizes, int n_in,
                              void* d_out, int out_size, void* d_ws, size_t ws_size,
                              hipStream_t stream) {
    const int*   x          = (const int*)d_in[0];
    const int*   edge_index = (const int*)d_in[1];
    // d_in[2] = edge_attr (unused by reference path)
    const int*   batch      = (const int*)d_in[3];
    const float* emb_table  = (const float*)d_in[4];
    const float* W1         = (const float*)d_in[5];
    const float* b1         = (const float*)d_in[6];
    const float* W2         = (const float*)d_in[7];
    const float* b2         = (const float*)d_in[8];
    const float* linW       = (const float*)d_in[9];
    const float* linb       = (const float*)d_in[10];
    float*       out        = (float*)d_out;

    const int N = in_sizes[0];          // 100000 (multiple of 16)
    const int E = in_sizes[1] / 2;      // 1250000
    const int* esrc = edge_index;       // row 0
    const int* edst = edge_index + E;   // row 1

    // workspace: deg[N] f32, dinv[N] f32, B/C = [N,64] f32, Abf = [N,64] bf16,
    //            Wp1/Wp2 = 4096 bf16 each  (~65 MB total; all chunks 32B-aligned)
    float*  deg  = (float*)d_ws;
    float*  dinv = deg + N;
    float*  B    = dinv + N;                       // hs (messages)
    float*  C    = B + (size_t)N * FDIM;           // acc (aggregation)
    __bf16* Abf  = (__bf16*)(C + (size_t)N * FDIM);
    __bf16* Wp1  = Abf + (size_t)N * FDIM;
    __bf16* Wp2  = Wp1 + 4096;

    const int TPB = 256;
    const long long nodeFeats = (long long)N * FDIM;
    const long long edgeFeats = (long long)E * FDIM;
    const int nTiles     = N / 16;
    const int gemmBlocks = (nTiles + 7) / 8;                 // 8 waves per block
    const int nfBlocks   = (int)((nodeFeats + TPB - 1) / TPB);
    const int efBlocks   = (int)((edgeFeats + TPB - 1) / TPB);

    // weight pre-pack + degree (with self-loops) + embedding gather (bf16)
    pack_w_kernel<<<8192 / TPB, TPB, 0, stream>>>(W1, W2, Wp1, Wp2);
    init_deg_kernel<<<(N + TPB - 1) / TPB, TPB, 0, stream>>>(deg, N);
    deg_accum_kernel<<<(E + TPB - 1) / TPB, TPB, 0, stream>>>(edst, deg, E);
    embed_kernel<<<nfBlocks, TPB, 0, stream>>>(x, emb_table, deg, Abf, dinv, N);

    // layer 1: B = (Abf@W1)*dinv, C = B (self-loop init); scatter; relu -> Abf (bf16)
    gemm64_scale_kernel<<<gemmBlocks, TPB, 0, stream>>>(Abf, Wp1, dinv, B, C, nTiles);
    scatter_kernel<<<efBlocks, TPB, 0, stream>>>(esrc, edst, B, C, edgeFeats);
    finalize_bf16_kernel<<<nfBlocks, TPB, 0, stream>>>(C, dinv, b1, Abf, N);

    // layer 2: B = (Abf@W2)*dinv, C = B; scatter; finalize (f32) -> B
    gemm64_scale_kernel<<<gemmBlocks, TPB, 0, stream>>>(Abf, Wp2, dinv, B, C, nTiles);
    scatter_kernel<<<efBlocks, TPB, 0, stream>>>(esrc, edst, B, C, edgeFeats);
    finalize_f32_kernel<<<nfBlocks, TPB, 0, stream>>>(C, dinv, b2, B, N);

    // readout
    zero_out_kernel<<<(out_size + TPB - 1) / TPB, TPB, 0, stream>>>(out, out_size);
    readout_kernel<<<(int)(((long long)N * 32 + TPB - 1) / TPB), TPB, 0, stream>>>(
        B, linW, linb, batch, out, N);
}
